// MultiHeadAttention_36438502539439
// MI455X (gfx1250) — matrix operand order repaired and checked by reading, not compiled, not run on hardware
//
#include <hip/hip_runtime.h>
#include <hip/hip_bf16.h>
#include <stdint.h>

// MHA: B=4, S=2048, E=1024, H=16, D=64. All matmuls via v_wmma_f32_16x16x32_bf16.

typedef __attribute__((ext_vector_type(16))) __bf16 v16bf;
typedef __attribute__((ext_vector_type(8)))  float  v8f;

#define EDIM 1024
#define HEADS 16
#define HD 64
#define BATCH 4
#define SEQ 2048
#define MROWS (BATCH * SEQ)

static __device__ __forceinline__ uint16_t f2bf(float f) {
    union { float f; uint32_t u; } v; v.f = f;
    uint32_t u = v.u;
    u += 0x7FFFu + ((u >> 16) & 1u);   // round-to-nearest-even
    return (uint16_t)(u >> 16);
}
static __device__ __forceinline__ uint32_t pack2(float a, float b) {
    return (uint32_t)f2bf(a) | ((uint32_t)f2bf(b) << 16);
}

union AFrag {           // 16 bf16 = 8 VGPRs
    v16bf v;
    uint32_t u[8];
    uint4 q[2];
};

// Async global->LDS copy of 16 bytes (ASYNCcnt-tracked, gfx1250).
static __device__ __forceinline__ void async_b128(uint32_t lds_addr, const void* gaddr) {
    asm volatile("global_load_async_to_lds_b128 %0, %1, off"
                 :: "v"(lds_addr), "v"(gaddr) : "memory");
}
static __device__ __forceinline__ void wait_asynccnt0() {
    asm volatile("s_wait_asynccnt 0" ::: "memory");
}

// ---------------------------------------------------------------------------
// GEMM: out = X (M x K) * W^T (W is N x K row-major) + bias
// OUT_MODE 0: bf16 out, (B,H,S,D) layout   (q, k)
// OUT_MODE 1: bf16 out, (B,H,D,S) layout   (v transposed)
// OUT_MODE 2: f32  out, row-major M x N    (final projection)
// Block: 256 thr = 8 waves, tile 128(M) x 64(N), BK = 32.
// Wave w computes 32x32: rows (w&3)*32, cols (w>>2)*32 -> 2x2 WMMA tiles.
// LDS layouts (bf16, row stride 40 halves = 80B, keeps b128 alignment):
//   As[row][k]  : A fragment lane(m=l%16,kg=l/16): halves at k = kg*16+{0..15}
//   Bs[col][k]  : B fragment lane(n=l%16,kg=l/16): halves at k = kg*16+{0..15}
// IN_BF16 path stages A with GLOBAL_LOAD_ASYNC_TO_LDS_B128.
// ---------------------------------------------------------------------------
template <int OUT_MODE, bool IN_BF16>
__global__ __launch_bounds__(256) void gemm_xwt(const void* __restrict__ Xv,
                                                const float* __restrict__ W,
                                                const float* __restrict__ bias,
                                                void* __restrict__ outv) {
    __shared__ uint16_t As[128 * 40];
    __shared__ uint16_t Bs[64 * 40];

    const int tid  = threadIdx.x;
    const int lane = tid & 31;
    const int w    = tid >> 5;
    const int m_blk = blockIdx.y * 128;
    const int n_blk = blockIdx.x * 64;
    const int wm = (w & 3) * 32;
    const int wn = (w >> 2) * 32;
    const int lm = lane & 15;
    const int kg = lane >> 4;

    const int arow  = tid >> 1;          // 0..127
    const int akoff = (tid & 1) * 16;
    const int bcol  = (tid >> 1) & 63;   // 0..63 (used when tid < 128)
    const int bkoff = (tid & 1) * 16;

    v8f acc[2][2] = {};

    for (int k0 = 0; k0 < EDIM; k0 += 32) {
        // ---- stage A tile (128 x 32) ----
        if (IN_BF16) {
            // raw bf16 copy: async global->LDS, no VGPR round trip
            const uint16_t* p = (const uint16_t*)Xv +
                                (size_t)(m_blk + arow) * EDIM + k0 + akoff;
            const uint32_t lds = (uint32_t)(uintptr_t)&As[arow * 40 + akoff];
            async_b128(lds, p);
            async_b128(lds + 16, p + 8);
        } else {
            const float* p = (const float*)Xv +
                             (size_t)(m_blk + arow) * EDIM + k0 + akoff;
            float4 x0 = *(const float4*)(p + 0);
            float4 x1 = *(const float4*)(p + 4);
            float4 x2 = *(const float4*)(p + 8);
            float4 x3 = *(const float4*)(p + 12);
            uint4 lo = make_uint4(pack2(x0.x, x0.y), pack2(x0.z, x0.w),
                                  pack2(x1.x, x1.y), pack2(x1.z, x1.w));
            uint4 hi = make_uint4(pack2(x2.x, x2.y), pack2(x2.z, x2.w),
                                  pack2(x3.x, x3.y), pack2(x3.z, x3.w));
            *(uint4*)&As[arow * 40 + akoff]     = lo;
            *(uint4*)&As[arow * 40 + akoff + 8] = hi;
        }
        // ---- stage B tile (64 cols x 32 k) from W (row-major N x K) ----
        if (tid < 128) {
            const float* p = W + (size_t)(n_blk + bcol) * EDIM + k0 + bkoff;
            float4 x0 = *(const float4*)(p + 0);
            float4 x1 = *(const float4*)(p + 4);
            float4 x2 = *(const float4*)(p + 8);
            float4 x3 = *(const float4*)(p + 12);
            uint4 lo = make_uint4(pack2(x0.x, x0.y), pack2(x0.z, x0.w),
                                  pack2(x1.x, x1.y), pack2(x1.z, x1.w));
            uint4 hi = make_uint4(pack2(x2.x, x2.y), pack2(x2.z, x2.w),
                                  pack2(x3.x, x3.y), pack2(x3.z, x3.w));
            *(uint4*)&Bs[bcol * 40 + bkoff]     = lo;
            *(uint4*)&Bs[bcol * 40 + bkoff + 8] = hi;
        }
        // prefetch next K-slab (stay inside this block's tiles: in-bounds)
        if (k0 + 32 < EDIM) {
            const char* Xc = (const char*)Xv;
            const size_t esz = IN_BF16 ? 2 : 4;
            __builtin_prefetch(Xc + ((size_t)(m_blk + arow) * EDIM + k0 + 32 + akoff) * esz, 0, 1);
            if (tid < 128)
                __builtin_prefetch((const char*)(W + (size_t)(n_blk + bcol) * EDIM + k0 + 32 + bkoff), 0, 1);
        }
        if (IN_BF16) wait_asynccnt0();   // own async writes landed before barrier
        __syncthreads();

        AFrag a[2], b[2];
#pragma unroll
        for (int i = 0; i < 2; ++i) {
            const int base = (wm + i * 16 + lm) * 40 + kg * 16;
            a[i].q[0] = *(const uint4*)&As[base];
            a[i].q[1] = *(const uint4*)&As[base + 8];
        }
#pragma unroll
        for (int j = 0; j < 2; ++j) {
            const int base = (wn + j * 16 + lm) * 40 + kg * 16;
            b[j].q[0] = *(const uint4*)&Bs[base];
            b[j].q[1] = *(const uint4*)&Bs[base + 8];
        }
#pragma unroll
        for (int i = 0; i < 2; ++i)
#pragma unroll
            for (int j = 0; j < 2; ++j)
                acc[i][j] = __builtin_amdgcn_wmma_f32_16x16x32_bf16(
                    false, a[i].v, false, b[j].v, (short)0, acc[i][j], false, false);
        __syncthreads();
    }

    // ---- epilogue: C layout: vgpr r -> row r + kg*8, col = lm ----
#pragma unroll
    for (int j = 0; j < 2; ++j) {
        const int gcol = n_blk + wn + j * 16 + lm;
        const float bc = bias[gcol];
#pragma unroll
        for (int i = 0; i < 2; ++i) {
#pragma unroll
            for (int r = 0; r < 8; ++r) {
                const int grow = m_blk + wm + i * 16 + r + kg * 8;
                const float val = acc[i][j][r] + bc;
                if (OUT_MODE == 2) {
                    ((float*)outv)[(size_t)grow * EDIM + gcol] = val;
                } else {
                    const int b_ = grow >> 11, s_ = grow & (SEQ - 1);
                    const int h_ = gcol >> 6, d_ = gcol & (HD - 1);
                    size_t idx;
                    if (OUT_MODE == 0)      // (B,H,S,D)
                        idx = (((size_t)(b_ * HEADS + h_)) * SEQ + s_) * HD + d_;
                    else                    // (B,H,D,S)
                        idx = (((size_t)(b_ * HEADS + h_)) * HD + d_) * SEQ + s_;
                    ((uint16_t*)outv)[idx] = f2bf(val);
                }
            }
        }
    }
}

// ---------------------------------------------------------------------------
// Attention: one wave handles (b,h, 16-query tile) with flash-style online
// softmax over 32-key chunks. q/k: (B,H,S,D) bf16; v: (B,H,D,S) bf16.
// Output: bf16 (B,S,E) for the final projection.
// ---------------------------------------------------------------------------
__global__ __launch_bounds__(128) void attn_kernel(const uint16_t* __restrict__ qb,
                                                   const uint16_t* __restrict__ kb,
                                                   const uint16_t* __restrict__ vT,
                                                   uint16_t* __restrict__ ao) {
    __shared__ uint16_t Ps[4][16 * 40];   // per-wave probability re-layout scratch

    const int tid  = threadIdx.x;
    const int lane = tid & 31;
    const int wv   = tid >> 5;
    const int lm   = lane & 15;
    const int kg   = lane >> 4;
    const int wflat = blockIdx.x * 4 + wv;
    const int bh    = wflat >> 7;              // 0..63 = b*16+h
    const int qbase = (wflat & 127) * 16;

    const uint16_t* qp = qb + ((size_t)bh * SEQ + qbase) * HD;
    const uint16_t* kp = kb + (size_t)bh * SEQ * HD;
    const uint16_t* vp = vT + (size_t)bh * HD * SEQ;
    uint16_t* Psw = Ps[wv];

    // Q A-fragments (rows qbase..+15, d split 0..31 / 32..63), resident all loop
    AFrag aq[2];
#pragma unroll
    for (int ds = 0; ds < 2; ++ds) {
        const uint16_t* p = qp + (size_t)lm * HD + ds * 32 + kg * 16;
        aq[ds].q[0] = *(const uint4*)p;
        aq[ds].q[1] = *(const uint4*)(p + 8);
    }

    v8f accO[4] = {};
    float mrun[8], lrun[8];
#pragma unroll
    for (int r = 0; r < 8; ++r) { mrun[r] = -1e30f; lrun[r] = 0.0f; }

    for (int kc = 0; kc < SEQ; kc += 32) {
        // ---- scores: two 16x16 tiles over this 32-key chunk ----
        v8f sc[2];
#pragma unroll
        for (int j = 0; j < 2; ++j) {
            v8f s = {};
#pragma unroll
            for (int ds = 0; ds < 2; ++ds) {
                AFrag bk;
                const uint16_t* kr =
                    kp + (size_t)(kc + j * 16 + lm) * HD + ds * 32 + kg * 16;
                bk.q[0] = *(const uint4*)kr;
                bk.q[1] = *(const uint4*)(kr + 8);
                s = __builtin_amdgcn_wmma_f32_16x16x32_bf16(
                    false, aq[ds].v, false, bk.v, (short)0, s, false, false);
            }
            sc[j] = s;
        }
        if (kc + 32 < SEQ) {
            __builtin_prefetch(kp + (size_t)(kc + 32 + lm) * HD, 0, 1);
            __builtin_prefetch(vp + (size_t)lm * SEQ + kc + 32, 0, 1);
        }

        // ---- online softmax update (row r lives at C row r+kg*8, cols=lanes) ----
#pragma unroll
        for (int r = 0; r < 8; ++r) {
            float s0 = sc[0][r] * 0.125f;          // 1/sqrt(64)
            float s1 = sc[1][r] * 0.125f;
            float t = fmaxf(s0, s1);
#pragma unroll
            for (int off = 8; off > 0; off >>= 1)
                t = fmaxf(t, __shfl_xor(t, off, 32));
            const float mn   = fmaxf(mrun[r], t);
            const float corr = __expf(mrun[r] - mn);
            const float p0 = __expf(s0 - mn);
            const float p1 = __expf(s1 - mn);
            float rs = p0 + p1;
#pragma unroll
            for (int off = 8; off > 0; off >>= 1)
                rs += __shfl_xor(rs, off, 32);
            lrun[r] = lrun[r] * corr + rs;
            mrun[r] = mn;
#pragma unroll
            for (int j2 = 0; j2 < 4; ++j2) accO[j2][r] *= corr;
            const int row = r + kg * 8;
            Psw[row * 40 + lm]      = f2bf(p0);
            Psw[row * 40 + 16 + lm] = f2bf(p1);
        }
        asm volatile("s_wait_dscnt 0" ::: "memory");
        __builtin_amdgcn_wave_barrier();

        // ---- P re-layout to A-fragment, then accO += P * V ----
        AFrag pa;
        pa.q[0] = *(const uint4*)&Psw[lm * 40 + kg * 16];
        pa.q[1] = *(const uint4*)&Psw[lm * 40 + kg * 16 + 8];
#pragma unroll
        for (int j2 = 0; j2 < 4; ++j2) {
            AFrag bv;
            const uint16_t* vr =
                vp + (size_t)(j2 * 16 + lm) * SEQ + kc + kg * 16;
            bv.q[0] = *(const uint4*)vr;
            bv.q[1] = *(const uint4*)(vr + 8);
            accO[j2] = __builtin_amdgcn_wmma_f32_16x16x32_bf16(
                false, pa.v, false, bv.v, (short)0, accO[j2], false, false);
        }
    }

    // ---- epilogue: normalize and store bf16 (B,S,E) ----
    const int h_ = bh & 15, b_ = bh >> 4;
#pragma unroll
    for (int r = 0; r < 8; ++r) {
        const float inv = 1.0f / lrun[r];
        const int row = qbase + r + kg * 8;
        const size_t base = ((size_t)b_ * SEQ + row) * EDIM + h_ * HD;
#pragma unroll
        for (int j2 = 0; j2 < 4; ++j2)
            ao[base + j2 * 16 + lm] = f2bf(accO[j2][r] * inv);
    }
}

// ---------------------------------------------------------------------------
extern "C" void kernel_launch(void* const* d_in, const int* in_sizes, int n_in,
                              void* d_out, int out_size, void* d_ws, size_t ws_size,
                              hipStream_t stream) {
    const float* Q  = (const float*)d_in[0];
    const float* K  = (const float*)d_in[1];
    const float* V  = (const float*)d_in[2];
    const float* Wq = (const float*)d_in[3];
    const float* bq = (const float*)d_in[4];
    const float* Wk = (const float*)d_in[5];
    const float* bk = (const float*)d_in[6];
    const float* Wv = (const float*)d_in[7];
    const float* bv = (const float*)d_in[8];
    const float* Wo = (const float*)d_in[9];
    const float* bo = (const float*)d_in[10];

    // workspace: 4 bf16 buffers of MROWS*EDIM (16 MB each)
    uint16_t* qb = (uint16_t*)d_ws;
    uint16_t* kb = qb + (size_t)MROWS * EDIM;
    uint16_t* vT = kb + (size_t)MROWS * EDIM;
    uint16_t* ao = vT + (size_t)MROWS * EDIM;

    dim3 g(EDIM / 64, MROWS / 128);  // 16 x 64 blocks

    gemm_xwt<0, false><<<g, 256, 0, stream>>>(Q, Wq, bq, qb);   // q  (B,H,S,D)
    gemm_xwt<0, false><<<g, 256, 0, stream>>>(K, Wk, bk, kb);   // k  (B,H,S,D)
    gemm_xwt<1, false><<<g, 256, 0, stream>>>(V, Wv, bv, vT);   // v^T(B,H,D,S)

    // 64 bh * 128 q-tiles = 8192 waves / 4 waves-per-block = 2048 blocks
    attn_kernel<<<2048, 128, 0, stream>>>(qb, kb, vT, ao);

    gemm_xwt<2, true><<<g, 256, 0, stream>>>(ao, Wo, bo, d_out);
}
